// CompGCN_40097814675485
// MI455X (gfx1250) — matrix operand order repaired and checked by reading, not compiled, not run on hardware
//
#include <hip/hip_runtime.h>

typedef __attribute__((ext_vector_type(16))) _Float16 v16h;
typedef __attribute__((ext_vector_type(8)))  _Float16 v8h;
typedef __attribute__((ext_vector_type(8)))  float    v8f;

#define N_NODES 50000
#define N_EDGES 500000
#define FD 128
#define N_REL2 200
#define BN_EPS 1e-5f

// ---------------------------------------------------------------------------
// Pack a [128,128] f32 weight (einsum 'nd,od->no' => B[k=d, n=o] = W[o,d])
// into WMMA-B fragment order for v_wmma_f32_16x16x32_f16:
//   tile (kt in 0..3, nt in 0..7): per lane l, halves h=0..15:
//     n = nt*16 + (l&15), k = kt*32 + 8*((l>>4)&1) + (h&7) + 16*(h>>3)
// Flat: dst[((kt*8+nt)*32 + lane)*16 + h]
// ---------------------------------------------------------------------------
__global__ void compgcn_pack_w(const float* __restrict__ Wloop,
                               const float* __restrict__ Win,
                               const float* __restrict__ Wrel,
                               _Float16* __restrict__ dst) {
  int b = blockIdx.x;
  const float* src;
  switch (b) {
    case 0: src = Wloop;             break;  // layer0 W_loop
    case 1: src = Win;               break;  // layer0 W_in
    case 2: src = Wrel;              break;  // layer0 W_rel
    case 3: src = Wloop + 16384;     break;  // layer1 W_loop
    default: src = Win + 16384;      break;  // layer1 W_in
  }
  _Float16* out = dst + b * 16384;
  for (int t = threadIdx.x; t < 16384; t += blockDim.x) {
    int h    = t & 15;
    int lane = (t >> 4) & 31;
    int nt   = (t >> 9) & 7;
    int kt   = t >> 12;
    int o = nt * 16 + (lane & 15);
    int d = kt * 32 + 8 * ((lane >> 4) & 1) + (h & 7) + 16 * (h >> 3);
    out[t] = (_Float16)src[o * FD + d];
  }
}

// ---------------------------------------------------------------------------
// Shared WMMA core: one wave computes a 16x128 = A(16x128) @ B(128x128) tile.
// ldsRow: this wave's staged A rows, f16 [16][128].
// A-fragment layout (ISA 16-bit A 16x32): lane m=l&15; blk=(l>>4)&1;
//   halves 0..7  -> K = kc + 8*blk + 0..7
//   halves 8..15 -> K = kc + 16 + 8*blk + 0..7
// ---------------------------------------------------------------------------
__device__ __forceinline__ void compgcn_wmma_tile(const _Float16 (*ldsRow)[FD],
                                                  int lane,
                                                  const _Float16* __restrict__ wpack,
                                                  v8f acc[8]) {
  const int blk  = (lane >> 4) & 1;
  const int mrow = lane & 15;
#pragma unroll
  for (int kt = 0; kt < 4; ++kt) {
    const int kc = kt * 32;
    v8h lo = *(const v8h*)&ldsRow[mrow][kc + 8 * blk];
    v8h hi = *(const v8h*)&ldsRow[mrow][kc + 16 + 8 * blk];
    v16h a = __builtin_shufflevector(lo, hi, 0, 1, 2, 3, 4, 5, 6, 7,
                                     8, 9, 10, 11, 12, 13, 14, 15);
    const _Float16* wt = wpack + (kt * 8) * 512;
#pragma unroll
    for (int nt = 0; nt < 8; ++nt) {
      v16h bm = *(const v16h*)&wt[nt * 512 + lane * 16];
      acc[nt] = __builtin_amdgcn_wmma_f32_16x16x32_f16(
          false, a, false, bm, (short)0, acc[nt], false, false);
    }
  }
}

// ---------------------------------------------------------------------------
// dst[m,:] = rows(src, gather)[m,:] @ W^T + bias     (self-loop / relation GEMM)
// 128 threads = 4 waves; wave w owns rows [m0, m0+16).
// ---------------------------------------------------------------------------
__global__ __launch_bounds__(128)
void compgcn_gemm_rows(const float* __restrict__ src,
                       const int* __restrict__ gather,
                       int M,
                       const _Float16* __restrict__ wpack,
                       const float* __restrict__ bias,
                       float* __restrict__ dst) {
  __shared__ __align__(16) _Float16 ldsA[4][16][FD];
  const int tid  = threadIdx.x;
  const int wave = tid >> 5;
  const int lane = tid & 31;
  const int m0   = (blockIdx.x * 4 + wave) * 16;

  // Stage A rows (f32 -> f16), coalesced.
#pragma unroll 4
  for (int j = 0; j < 64; ++j) {
    int idx = j * 32 + lane;
    int ml = idx >> 7, d = idx & 127;
    int m = m0 + ml;
    float v = 0.0f;
    if (m < M) {
      int rr = gather ? gather[m] : m;
      v = src[(long)rr * FD + d];
    }
    ldsA[wave][ml][d] = (_Float16)v;
  }
  __syncthreads();

  v8f acc[8];
#pragma unroll
  for (int nt = 0; nt < 8; ++nt)
#pragma unroll
    for (int i = 0; i < 8; ++i) acc[nt][i] = 0.0f;

  compgcn_wmma_tile(ldsA[wave], lane, wpack, acc);

  const int blk = (lane >> 4) & 1;
#pragma unroll
  for (int nt = 0; nt < 8; ++nt) {
    int n = nt * 16 + (lane & 15);
    float bv = bias[n];
#pragma unroll
    for (int c = 0; c < 8; ++c) {
      int m = m0 + c + 8 * blk;
      if (m < M) dst[(long)m * FD + n] = acc[nt][c] + bv;
    }
  }
}

// ---------------------------------------------------------------------------
// Per-edge: msg = (x[col] * r[etype]) @ W_in^T + b_in ; atomic scatter to out[row].
// 128 threads = 4 waves; wave owns 16 edges.
// ---------------------------------------------------------------------------
__global__ __launch_bounds__(128)
void compgcn_edge_msg(const float* __restrict__ xsrc,
                      const int* __restrict__ gather,
                      const float* __restrict__ rel,
                      const int* __restrict__ eidx,   // [2][E]: row then col
                      const int* __restrict__ etype,
                      const _Float16* __restrict__ wpack,
                      const float* __restrict__ bias,
                      float* __restrict__ out) {
  __shared__ __align__(16) _Float16 ldsA[4][16][FD];
  const int tid  = threadIdx.x;
  const int wave = tid >> 5;
  const int lane = tid & 31;
  const int e0   = (blockIdx.x * 4 + wave) * 16;
  const int* col = eidx + N_EDGES;

  // Stage composed = x[col] * r[etype] (f32 -> f16); per j all lanes share one edge.
#pragma unroll 4
  for (int j = 0; j < 64; ++j) {
    int idx = j * 32 + lane;
    int ml = idx >> 7, d = idx & 127;
    int e = e0 + ml;
    float v = 0.0f;
    if (e < N_EDGES) {
      int c = col[e];
      int t = etype[e];
      int rr = gather ? gather[c] : c;
      v = xsrc[(long)rr * FD + d] * rel[(long)t * FD + d];
    }
    ldsA[wave][ml][d] = (_Float16)v;
  }
  __syncthreads();

  v8f acc[8];
#pragma unroll
  for (int nt = 0; nt < 8; ++nt)
#pragma unroll
    for (int i = 0; i < 8; ++i) acc[nt][i] = 0.0f;

  compgcn_wmma_tile(ldsA[wave], lane, wpack, acc);

  const int blk = (lane >> 4) & 1;
#pragma unroll
  for (int nt = 0; nt < 8; ++nt) {
    int n = nt * 16 + (lane & 15);
    float bv = bias[n];
#pragma unroll
    for (int c = 0; c < 8; ++c) {
      int e = e0 + c + 8 * blk;
      if (e < N_EDGES) {
        int rdst = eidx[e];  // row (destination node)
        unsafeAtomicAdd(&out[(long)rdst * FD + n], acc[nt][c] + bv);
      }
    }
  }
}

// ---------------------------------------------------------------------------
// BatchNorm: zero accumulators, partial column sums, fused normalize(+ReLU).
// ---------------------------------------------------------------------------
__global__ void compgcn_bn_init(float* __restrict__ sums) {
  sums[threadIdx.x] = 0.0f;  // 256 floats: [0..127]=sum, [128..255]=sumsq
}

__global__ __launch_bounds__(256)
void compgcn_bn_partial(const float* __restrict__ x, float* __restrict__ sums, int N) {
  __shared__ float ls[256];
  __shared__ float lss[256];
  const int tid = threadIdx.x;
  const int d = tid & 127, sub = tid >> 7;
  float s = 0.0f, ss = 0.0f;
  for (int n = blockIdx.x * 2 + sub; n < N; n += gridDim.x * 2) {
    float v = x[(long)n * FD + d];
    s += v;
    ss += v * v;
  }
  ls[tid] = s;
  lss[tid] = ss;
  __syncthreads();
  if (tid < 128) {
    unsafeAtomicAdd(&sums[d], ls[tid] + ls[tid + 128]);
    unsafeAtomicAdd(&sums[128 + d], lss[tid] + lss[tid + 128]);
  }
}

__global__ __launch_bounds__(256)
void compgcn_bn_apply(float* __restrict__ x, const float* __restrict__ sums,
                      const float* __restrict__ gamma, const float* __restrict__ beta,
                      int N, int relu) {
  const float invN = 1.0f / (float)N;
  long total = (long)N * FD;
  for (long idx = (long)blockIdx.x * blockDim.x + threadIdx.x; idx < total;
       idx += (long)gridDim.x * blockDim.x) {
    int d = (int)(idx & 127);
    float mean = sums[d] * invN;
    float var = sums[128 + d] * invN - mean * mean;
    float inv = rsqrtf(var + BN_EPS);
    float v = (x[idx] - mean) * inv * gamma[d] + beta[d];
    if (relu) v = fmaxf(v, 0.0f);
    x[idx] = v;
  }
}

// ---------------------------------------------------------------------------
extern "C" void kernel_launch(void* const* d_in, const int* in_sizes, int n_in,
                              void* d_out, int out_size, void* d_ws, size_t ws_size,
                              hipStream_t stream) {
  (void)in_sizes; (void)n_in; (void)out_size; (void)ws_size;
  const int*   entity_ids = (const int*)d_in[0];
  const int*   edge_index = (const int*)d_in[1];
  const int*   edge_type  = (const int*)d_in[2];
  const float* ent_emb    = (const float*)d_in[3];
  const float* rel_emb    = (const float*)d_in[4];
  const float* W_loop     = (const float*)d_in[5];
  const float* b_loop     = (const float*)d_in[6];
  const float* W_in       = (const float*)d_in[7];
  const float* b_in       = (const float*)d_in[8];
  const float* W_rel      = (const float*)d_in[9];
  const float* b_rel      = (const float*)d_in[10];
  const float* gamma      = (const float*)d_in[11];
  const float* beta       = (const float*)d_in[12];

  char* ws = (char*)d_ws;
  float*    X1   = (float*)ws;                      // 50000*128 f32 = 25,600,000 B
  float*    R1   = (float*)(ws + 25600000);         // 200*128 f32   =    102,400 B
  _Float16* WP   = (_Float16*)(ws + 25702400);      // 5*16384 f16   =    163,840 B
  float*    SUMS = (float*)(ws + 25866240);         // 256 f32
  float*    OUT  = (float*)d_out;

  const int gemm_blocks = (N_NODES + 63) / 64;      // 4 waves * 16 rows each
  const int edge_blocks = (N_EDGES + 63) / 64;

  // Pack all five weight matrices into WMMA-B f16 fragments.
  compgcn_pack_w<<<5, 256, 0, stream>>>(W_loop, W_in, W_rel, WP);

  // r1 = rel_emb @ W_rel0^T + b_rel0  (layer-1 relation update is unused output)
  compgcn_gemm_rows<<<(N_REL2 + 63) / 64, 128, 0, stream>>>(
      rel_emb, nullptr, N_REL2, WP + 2 * 16384, b_rel, R1);

  // ---- Layer 0 ----
  compgcn_gemm_rows<<<gemm_blocks, 128, 0, stream>>>(
      ent_emb, entity_ids, N_NODES, WP + 0 * 16384, b_loop, X1);
  compgcn_edge_msg<<<edge_blocks, 128, 0, stream>>>(
      ent_emb, entity_ids, rel_emb, edge_index, edge_type,
      WP + 1 * 16384, b_in, X1);
  compgcn_bn_init<<<1, 256, 0, stream>>>(SUMS);
  compgcn_bn_partial<<<256, 256, 0, stream>>>(X1, SUMS, N_NODES);
  compgcn_bn_apply<<<512, 256, 0, stream>>>(X1, SUMS, gamma, beta, N_NODES, 1);

  // ---- Layer 1 (writes straight into d_out) ----
  compgcn_gemm_rows<<<gemm_blocks, 128, 0, stream>>>(
      X1, nullptr, N_NODES, WP + 3 * 16384, b_loop + FD, OUT);
  compgcn_edge_msg<<<edge_blocks, 128, 0, stream>>>(
      X1, nullptr, R1, edge_index, edge_type, WP + 4 * 16384, b_in + FD, OUT);
  compgcn_bn_init<<<1, 256, 0, stream>>>(SUMS);
  compgcn_bn_partial<<<256, 256, 0, stream>>>(OUT, SUMS, N_NODES);
  compgcn_bn_apply<<<512, 256, 0, stream>>>(OUT, SUMS, gamma + FD, beta + FD, N_NODES, 0);
}
